// GatedAttentionUnit_76742475645156
// MI455X (gfx1250) — matrix operand order repaired and checked by reading, not compiled, not run on hardware
//
#include <hip/hip_runtime.h>
#include <hip/hip_fp16.h>

typedef __attribute__((ext_vector_type(16))) _Float16 v16h;
typedef __attribute__((ext_vector_type(8)))  _Float16 v8h;
typedef __attribute__((ext_vector_type(4)))  _Float16 v4h;
typedef __attribute__((ext_vector_type(8)))  float    v8f;

// Problem constants (match reference)
constexpr int B_  = 4;
constexpr int S_  = 2048;
constexpr int H_  = 768;
constexpr int I_  = 1536;
constexpr int DK_ = 128;
constexpr int NC_ = 2 * I_ + DK_;   // 3200
constexpr int BS_ = B_ * S_;        // 8192

__device__ __forceinline__ v8f wmma16(v16h a, v16h b, v8f c) {
  return __builtin_amdgcn_wmma_f32_16x16x32_f16(false, a, false, b, (short)0, c,
                                                false, false);
}

// A-matrix fragment (16x32 slab, 16-bit, row-major, leading dim ldm halves).
// lane L -> row M = mrow + (L&15); lanes 0-15: K={0..7,16..23}, lanes 16-31:
// K={8..15,24..31}. Two contiguous 16B chunks -> 2x ds_load_b128.
__device__ __forceinline__ v16h frag_A(const _Float16* base, int ldm, int mrow) {
  const int lane = threadIdx.x & 31;
  const _Float16* p =
      base + (size_t)(mrow + (lane & 15)) * ldm + ((lane >> 4) << 3);
  v8h lo = *(const v8h*)(p);
  v8h hi = *(const v8h*)(p + 16);
  return __builtin_shufflevector(lo, hi, 0, 1, 2, 3, 4, 5, 6, 7, 8, 9, 10, 11,
                                 12, 13, 14, 15);
}

// B-matrix fragment (32x16 slab) from *transposed* (n-major) LDS tile
// Bt[n][k], leading dim ldk halves. lane L -> col N = ncol+(L&15);
// lanes 0-15: K={0..7,16..23}, lanes 16-31: K={8..15,24..31}.
__device__ __forceinline__ v16h frag_Bt(const _Float16* base, int ldk, int ncol) {
  const int lane = threadIdx.x & 31;
  const _Float16* p =
      base + (size_t)(ncol + (lane & 15)) * ldk + ((lane >> 4) << 3);
  v8h lo = *(const v8h*)(p);
  v8h hi = *(const v8h*)(p + 16);
  return __builtin_shufflevector(lo, hi, 0, 1, 2, 3, 4, 5, 6, 7, 8, 9, 10, 11,
                                 12, 13, 14, 15);
}

// ---------------------------------------------------------------------------
// Kernel 1: X = silu(HS @ Wi); scatter into u16 (f16), v16 (f16), qkraw (f32)
// Tile: 128(M) x 64(N), K-step 32. 8 waves: 4x2 grid of 32x32 wave tiles.
// ---------------------------------------------------------------------------
__global__ __launch_bounds__(256) void gau_gemm1(const float* __restrict__ hs,
                                                 const float* __restrict__ Wi,
                                                 _Float16* __restrict__ u16,
                                                 _Float16* __restrict__ v16,
                                                 float* __restrict__ qkraw) {
  constexpr int LDA = 40, LDBT = 40;
  __shared__ _Float16 Ash[128 * LDA];     // [m][k]
  __shared__ _Float16 Bsh[64 * LDBT];     // [n][k] (transposed)
  const int n0 = blockIdx.x * 64;
  const int m0 = blockIdx.y * 128;
  const int w = threadIdx.x >> 5;
  const int wm = w >> 1, wn = w & 1;
  const int lane = threadIdx.x & 31;
  const int hi = lane >> 4, ln = lane & 15;

  v8f acc[2][2] = {};
  for (int k0 = 0; k0 < H_; k0 += 32) {
    __syncthreads();
    // A: 128x32 f32 -> f16, float4 reads, v4h packed LDS stores
    for (int i = threadIdx.x; i < 128 * 8; i += 256) {
      int r = i >> 3, c4 = (i & 7) << 2;
      float4 f = *(const float4*)&hs[(size_t)(m0 + r) * H_ + k0 + c4];
      v4h pk = {(_Float16)f.x, (_Float16)f.y, (_Float16)f.z, (_Float16)f.w};
      *(v4h*)&Ash[r * LDA + c4] = pk;
    }
    // B: 32x64 f32 -> transposed f16 tile; lane-coalesced along n,
    // packed 8B stores along k
    for (int i = threadIdx.x; i < 64 * 8; i += 256) {
      int n = i & 63, k4 = (i >> 6) << 2;
      const float* src = &Wi[(size_t)(k0 + k4) * NC_ + n0 + n];
      v4h pk = {(_Float16)src[0], (_Float16)src[NC_], (_Float16)src[2 * NC_],
                (_Float16)src[3 * NC_]};
      *(v4h*)&Bsh[n * LDBT + k4] = pk;
    }
    if (k0 + 32 < H_) {
      __builtin_prefetch(&hs[(size_t)(m0 + (threadIdx.x >> 1)) * H_ + k0 + 32], 0, 0);
      __builtin_prefetch(&Wi[(size_t)(k0 + 32 + (threadIdx.x >> 3)) * NC_ + n0], 0, 0);
    }
    __syncthreads();
    v16h a0 = frag_A(Ash, LDA, wm * 32);
    v16h a1 = frag_A(Ash, LDA, wm * 32 + 16);
    v16h b0 = frag_Bt(Bsh, LDBT, wn * 32);
    v16h b1 = frag_Bt(Bsh, LDBT, wn * 32 + 16);
    acc[0][0] = wmma16(a0, b0, acc[0][0]);
    acc[0][1] = wmma16(a0, b1, acc[0][1]);
    acc[1][0] = wmma16(a1, b0, acc[1][0]);
    acc[1][1] = wmma16(a1, b1, acc[1][1]);
  }
#pragma unroll
  for (int sm = 0; sm < 2; ++sm)
#pragma unroll
    for (int sn = 0; sn < 2; ++sn) {
      const int n = n0 + wn * 32 + sn * 16 + ln;
#pragma unroll
      for (int r = 0; r < 8; ++r) {
        const int m = m0 + wm * 32 + sm * 16 + r + hi * 8;
        float x = acc[sm][sn][r];
        float s = x / (1.0f + __expf(-x));  // silu
        if (n < I_)
          u16[(size_t)m * I_ + n] = (_Float16)s;
        else if (n < 2 * I_)
          v16[(size_t)m * I_ + (n - I_)] = (_Float16)s;
        else
          qkraw[(size_t)m * DK_ + (n - 2 * I_)] = s;
      }
    }
}

// ---------------------------------------------------------------------------
// Kernel 2: per-batch softmax scale = log(sum(mask)) / log(512) / sqrt(DK)
// ---------------------------------------------------------------------------
__global__ __launch_bounds__(256) void gau_scales(const int* __restrict__ mask,
                                                  float* __restrict__ scales) {
  __shared__ int red[256];
  const int b = blockIdx.x, t = threadIdx.x;
  int s = 0;
  for (int i = t; i < S_; i += 256) s += mask[b * S_ + i];
  red[t] = s;
  __syncthreads();
  for (int k = 128; k > 0; k >>= 1) {
    if (t < k) red[t] += red[t + k];
    __syncthreads();
  }
  if (t == 0) {
    float l = (float)red[0];
    scales[b] = (logf(l) / 6.2383246250395077f) * 0.08838834764831845f;
  }
}

// ---------------------------------------------------------------------------
// Kernel 3: rotary embedding: q16/k16 = rope(qk * w + b)
// ---------------------------------------------------------------------------
__global__ __launch_bounds__(256) void gau_rotary(
    const float* __restrict__ qkraw, const float* __restrict__ sinT,
    const float* __restrict__ cosT, const float* __restrict__ q_w,
    const float* __restrict__ q_b, const float* __restrict__ k_w,
    const float* __restrict__ k_b, _Float16* __restrict__ q16,
    _Float16* __restrict__ k16) {
  const int t = threadIdx.x;
  const int row = blockIdx.x * 64 + (t >> 2);
  const int j0 = (t & 3) * 16;
  const int m = row & (S_ - 1);
  const float* src = qkraw + (size_t)row * DK_;
  _Float16* qd = q16 + (size_t)row * DK_;
  _Float16* kd = k16 + (size_t)row * DK_;
#pragma unroll
  for (int j = j0; j < j0 + 16; ++j) {
    float x1 = src[2 * j], x2 = src[2 * j + 1];
    float sn = sinT[m * 64 + j], cs = cosT[m * 64 + j];
    float q1 = x1 * q_w[2 * j] + q_b[2 * j];
    float q2 = x2 * q_w[2 * j + 1] + q_b[2 * j + 1];
    float k1 = x1 * k_w[2 * j] + k_b[2 * j];
    float k2 = x2 * k_w[2 * j + 1] + k_b[2 * j + 1];
    qd[j] = (_Float16)(q1 * cs - q2 * sn);
    qd[j + 64] = (_Float16)(q1 * sn + q2 * cs);
    kd[j] = (_Float16)(k1 * cs - k2 * sn);
    kd[j + 64] = (_Float16)(k1 * sn + k2 * cs);
  }
}

// ---------------------------------------------------------------------------
// Kernel 4: attention probabilities. Full-row softmax (mask-aware), then the
// reference's post-softmax causal overwrite A[n>m] = -1e4. Writes A16 (f16).
// 128 query rows per block (1 wave = 16 rows); 16-key tiles staged n-major in
// LDS; scores via 4x v_wmma_f32_16x16x32_f16 (K=128).
// ---------------------------------------------------------------------------
__global__ __launch_bounds__(256) void gau_attn(const _Float16* __restrict__ q16,
                                                const _Float16* __restrict__ k16,
                                                const int* __restrict__ mask,
                                                const float* __restrict__ scales,
                                                _Float16* __restrict__ A16) {
  constexpr int LDKT = 136;  // 272B rows: 16B-aligned, bank-spread (68 % 64 = 4)
  __shared__ _Float16 Ksh[16 * LDKT];  // [n][d]
  const int b = blockIdx.y;
  const int m0 = blockIdx.x * 128;
  const int w = threadIdx.x >> 5;
  const int lane = threadIdx.x & 31;
  const int hi = lane >> 4, ln = lane & 15;
  const float scal = scales[b];
  const int mrow = m0 + w * 16;

  v16h qf[4];
  const _Float16* qbase = q16 + ((size_t)b * S_ + mrow) * DK_;
#pragma unroll
  for (int kk = 0; kk < 4; ++kk) qf[kk] = frag_A(qbase + kk * 32, DK_, 0);

  float rmax[8], rsum[8];
#pragma unroll
  for (int r = 0; r < 8; ++r) { rmax[r] = -3.0e38f; rsum[r] = 0.0f; }

  // pass 1: running max / sum-exp over ALL keys (reference softmaxes the full
  // row; causal is applied after softmax). pass 2: recompute + store.
  for (int pass = 0; pass < 2; ++pass) {
    for (int n0 = 0; n0 < S_; n0 += 16) {
      __syncthreads();
      {  // 16x128 f16 tile: one 16B chunk per thread, b128 -> b128 copy
        int n = threadIdx.x >> 4, d8 = (threadIdx.x & 15) << 3;
        *(v8h*)&Ksh[n * LDKT + d8] =
            *(const v8h*)&k16[((size_t)b * S_ + n0 + n) * DK_ + d8];
      }
      __syncthreads();
      v8f acc = {};
#pragma unroll
      for (int kk = 0; kk < 4; ++kk)
        acc = wmma16(qf[kk], frag_Bt(Ksh + kk * 32, LDKT, 0), acc);
      const int n = n0 + ln;
      const int mv = mask[b * S_ + n];
      if (pass == 0) {
#pragma unroll
        for (int r = 0; r < 8; ++r) {
          float s = mv ? acc[r] * scal : -10000.0f;
          float tmax = s;
#pragma unroll
          for (int xm = 1; xm < 16; xm <<= 1)
            tmax = fmaxf(tmax, __shfl_xor(tmax, xm, 32));
          float tsum = __expf(s - tmax);
#pragma unroll
          for (int xm = 1; xm < 16; xm <<= 1) tsum += __shfl_xor(tsum, xm, 32);
          float nm = fmaxf(rmax[r], tmax);
          rsum[r] = rsum[r] * __expf(rmax[r] - nm) + tsum * __expf(tmax - nm);
          rmax[r] = nm;
        }
      } else {
#pragma unroll
        for (int r = 0; r < 8; ++r) {
          const int m = mrow + r + hi * 8;
          float s = mv ? acc[r] * scal : -10000.0f;
          float val = (n > m) ? -10000.0f : __expf(s - rmax[r]) / rsum[r];
          A16[((size_t)b * S_ + m) * S_ + n] = (_Float16)val;
        }
      }
    }
  }
}

// ---------------------------------------------------------------------------
// Kernel 5: per-batch g = u ⊙ (A @ v).  M=S, N=I, K=S. Tile 128x64, K-step 32.
// ---------------------------------------------------------------------------
__global__ __launch_bounds__(256) void gau_av(const _Float16* __restrict__ A16,
                                              const _Float16* __restrict__ v16,
                                              const _Float16* __restrict__ u16,
                                              _Float16* __restrict__ g16) {
  constexpr int LDA = 40, LDBT = 40;
  __shared__ _Float16 Ash[128 * LDA];
  __shared__ _Float16 Bsh[64 * LDBT];
  const int b = blockIdx.z;
  const int n0 = blockIdx.x * 64;
  const int m0 = blockIdx.y * 128;
  const int w = threadIdx.x >> 5;
  const int wm = w >> 1, wn = w & 1;
  const int lane = threadIdx.x & 31;
  const int hi = lane >> 4, ln = lane & 15;
  const _Float16* Ab = A16 + (size_t)b * S_ * S_;

  v8f acc[2][2] = {};
  for (int k0 = 0; k0 < S_; k0 += 32) {
    __syncthreads();
    // A: 128x32 f16, b128 -> b128 copy
    for (int i = threadIdx.x; i < 128 * 4; i += 256) {
      int r = i >> 2, c8 = (i & 3) << 3;
      *(v8h*)&Ash[r * LDA + c8] =
          *(const v8h*)&Ab[(size_t)(m0 + r) * S_ + k0 + c8];
    }
    // B: 32x64 f16 -> transposed; lane-coalesced along n
    for (int i = threadIdx.x; i < 64 * 8; i += 256) {
      int n = i & 63, k4 = (i >> 6) << 2;
      const _Float16* src = &v16[((size_t)b * S_ + k0 + k4) * I_ + n0 + n];
      v4h pk = {src[0], src[I_], src[2 * I_], src[3 * I_]};
      *(v4h*)&Bsh[n * LDBT + k4] = pk;
    }
    if (k0 + 32 < S_) {
      __builtin_prefetch(&Ab[(size_t)(m0 + (threadIdx.x >> 1)) * S_ + k0 + 32], 0, 0);
      __builtin_prefetch(&v16[((size_t)b * S_ + k0 + 32 + (threadIdx.x >> 3)) * I_ + n0], 0, 0);
    }
    __syncthreads();
    v16h a0 = frag_A(Ash, LDA, wm * 32);
    v16h a1 = frag_A(Ash, LDA, wm * 32 + 16);
    v16h b0 = frag_Bt(Bsh, LDBT, wn * 32);
    v16h b1 = frag_Bt(Bsh, LDBT, wn * 32 + 16);
    acc[0][0] = wmma16(a0, b0, acc[0][0]);
    acc[0][1] = wmma16(a0, b1, acc[0][1]);
    acc[1][0] = wmma16(a1, b0, acc[1][0]);
    acc[1][1] = wmma16(a1, b1, acc[1][1]);
  }
#pragma unroll
  for (int sm = 0; sm < 2; ++sm)
#pragma unroll
    for (int sn = 0; sn < 2; ++sn) {
      const int n = n0 + wn * 32 + sn * 16 + ln;
#pragma unroll
      for (int r = 0; r < 8; ++r) {
        const int m = m0 + wm * 32 + sm * 16 + r + hi * 8;
        const size_t row = (size_t)b * S_ + m;
        float g = acc[sm][sn][r] * (float)u16[row * I_ + n];
        g16[row * I_ + n] = (_Float16)g;
      }
    }
}

// ---------------------------------------------------------------------------
// Kernel 6: out = g16 @ Wo   (M=BS, N=H, K=I), fp32 output.
// ---------------------------------------------------------------------------
__global__ __launch_bounds__(256) void gau_gemm2(const _Float16* __restrict__ g16,
                                                 const float* __restrict__ Wo,
                                                 float* __restrict__ out) {
  constexpr int LDA = 40, LDBT = 40;
  __shared__ _Float16 Ash[128 * LDA];
  __shared__ _Float16 Bsh[64 * LDBT];
  const int n0 = blockIdx.x * 64;
  const int m0 = blockIdx.y * 128;
  const int w = threadIdx.x >> 5;
  const int wm = w >> 1, wn = w & 1;
  const int lane = threadIdx.x & 31;
  const int hi = lane >> 4, ln = lane & 15;

  v8f acc[2][2] = {};
  for (int k0 = 0; k0 < I_; k0 += 32) {
    __syncthreads();
    for (int i = threadIdx.x; i < 128 * 4; i += 256) {
      int r = i >> 2, c8 = (i & 3) << 3;
      *(v8h*)&Ash[r * LDA + c8] =
          *(const v8h*)&g16[(size_t)(m0 + r) * I_ + k0 + c8];
    }
    for (int i = threadIdx.x; i < 64 * 8; i += 256) {
      int n = i & 63, k4 = (i >> 6) << 2;
      const float* src = &Wo[(size_t)(k0 + k4) * H_ + n0 + n];
      v4h pk = {(_Float16)src[0], (_Float16)src[H_], (_Float16)src[2 * H_],
                (_Float16)src[3 * H_]};
      *(v4h*)&Bsh[n * LDBT + k4] = pk;
    }
    if (k0 + 32 < I_) {
      __builtin_prefetch(&g16[(size_t)(m0 + (threadIdx.x >> 1)) * I_ + k0 + 32], 0, 0);
      __builtin_prefetch(&Wo[(size_t)(k0 + 32 + (threadIdx.x >> 3)) * H_ + n0], 0, 0);
    }
    __syncthreads();
    v16h a0 = frag_A(Ash, LDA, wm * 32);
    v16h a1 = frag_A(Ash, LDA, wm * 32 + 16);
    v16h b0 = frag_Bt(Bsh, LDBT, wn * 32);
    v16h b1 = frag_Bt(Bsh, LDBT, wn * 32 + 16);
    acc[0][0] = wmma16(a0, b0, acc[0][0]);
    acc[0][1] = wmma16(a0, b1, acc[0][1]);
    acc[1][0] = wmma16(a1, b0, acc[1][0]);
    acc[1][1] = wmma16(a1, b1, acc[1][1]);
  }
#pragma unroll
  for (int sm = 0; sm < 2; ++sm)
#pragma unroll
    for (int sn = 0; sn < 2; ++sn) {
      const int n = n0 + wn * 32 + sn * 16 + ln;
#pragma unroll
      for (int r = 0; r < 8; ++r) {
        const int m = m0 + wm * 32 + sm * 16 + r + hi * 8;
        out[(size_t)m * H_ + n] = acc[sm][sn][r];
      }
    }
}

// ---------------------------------------------------------------------------
extern "C" void kernel_launch(void* const* d_in, const int* in_sizes, int n_in,
                              void* d_out, int out_size, void* d_ws,
                              size_t ws_size, hipStream_t stream) {
  (void)in_sizes; (void)n_in; (void)out_size; (void)ws_size;
  const float* hs   = (const float*)d_in[0];
  const int*   mask = (const int*)d_in[1];
  const float* sinT = (const float*)d_in[2];
  const float* cosT = (const float*)d_in[3];
  const float* Wi   = (const float*)d_in[4];
  const float* Wo   = (const float*)d_in[5];
  const float* q_w  = (const float*)d_in[6];
  const float* q_b  = (const float*)d_in[7];
  const float* k_w  = (const float*)d_in[8];
  const float* k_b  = (const float*)d_in[9];
  float* out = (float*)d_out;

  char* ws = (char*)d_ws;
  size_t off = 0;
  auto alloc = [&](size_t bytes) -> char* {
    char* p = ws + off;
    off += (bytes + 255) & ~(size_t)255;
    return p;
  };
  _Float16* u16   = (_Float16*)alloc((size_t)BS_ * I_ * 2);
  _Float16* v16   = (_Float16*)alloc((size_t)BS_ * I_ * 2);
  _Float16* g16   = (_Float16*)alloc((size_t)BS_ * I_ * 2);
  _Float16* q16   = (_Float16*)alloc((size_t)BS_ * DK_ * 2);
  _Float16* k16   = (_Float16*)alloc((size_t)BS_ * DK_ * 2);
  float*    qkraw = (float*)alloc((size_t)BS_ * DK_ * 4);
  _Float16* A16   = (_Float16*)alloc((size_t)B_ * S_ * S_ * 2);
  float*    scl   = (float*)alloc(B_ * 4);

  gau_gemm1<<<dim3(NC_ / 64, BS_ / 128), 256, 0, stream>>>(hs, Wi, u16, v16, qkraw);
  gau_scales<<<B_, 256, 0, stream>>>(mask, scl);
  gau_rotary<<<BS_ / 64, 256, 0, stream>>>(qkraw, sinT, cosT, q_w, q_b, k_w, k_b,
                                           q16, k16);
  gau_attn<<<dim3(S_ / 128, B_), 256, 0, stream>>>(q16, k16, mask, scl, A16);
  gau_av<<<dim3(I_ / 64, S_ / 128, B_), 256, 0, stream>>>(A16, v16, u16, g16);
  gau_gemm2<<<dim3(H_ / 64, BS_ / 128), 256, 0, stream>>>(g16, Wo, out);
}
